// UnifiedGNNDotProduct_44057774523059
// MI455X (gfx1250) — compile-verified, matrix operand order checked by reading, not run on hardware
//
#include <hip/hip_runtime.h>
#include <math.h>

#define HID 512
#define HEADS 4
#define CH 128
#define DOT 256
#define F_IN 384
#define NEG_SLOPE 0.2f

typedef __attribute__((ext_vector_type(8)))  float          v8f;
typedef __attribute__((ext_vector_type(16))) __bf16         v16bf;
typedef __attribute__((ext_vector_type(4)))  unsigned int   v4u;
typedef __attribute__((ext_vector_type(8)))  int            v8i;
typedef __attribute__((ext_vector_type(4)))  int            v4i;

union Frag { v4u u[2]; v16bf v; };

__device__ __forceinline__ unsigned short f2bf(float f) {
  unsigned int u = __float_as_uint(f);
  unsigned int r = u + 0x7FFFu + ((u >> 16) & 1u);   // round-to-nearest-even
  return (unsigned short)(r >> 16);
}

// flags for GEMM epilogue
#define GF_RELU  1
#define GF_ACCUM 2
#define GF_OUTBF 4

#define BM 128
#define BN 128
#define BK 64

#if defined(__has_builtin)
#if __has_builtin(__builtin_amdgcn_tensor_load_to_lds) && __has_builtin(__builtin_amdgcn_s_wait_tensorcnt)
#define HAVE_TDM 1
#endif
#endif

#ifdef HAVE_TDM
// Issue one 2D TDM tile load: tile_dim1 rows x tile_dim0 elems of 2-byte data,
// row stride = stride0 elems, OOB (dim0/dim1) reads return zero.
// D# bitfields per CDNA5 ISA 8.3/8.4 (group0: count/lds/global/type, group1:
// data_size, tensor_dim0/1, tile_dim0/1, tensor_dim0_stride). Groups 2/3/pad zero.
// clang-23 builtin arity: (uint32x4, int32x8, int32x4, int32x4, int32x8, i32 cpol).
__device__ __forceinline__ void tdm_load_2d(unsigned lds_addr, const void* gptr,
                                            unsigned dim0, unsigned dim1,
                                            unsigned tile0, unsigned tile1,
                                            unsigned stride0)
{
  unsigned long long ga = (unsigned long long)gptr;
  v4u g0;
  g0[0] = 1u;                                        // count=1, user descriptor
  g0[1] = lds_addr;                                  // LDS byte address
  g0[2] = (unsigned)(ga & 0xFFFFFFFFull);            // global_addr[31:0]
  g0[3] = (unsigned)((ga >> 32) & 0x01FFFFFFull) | (2u << 30);  // addr[56:32], type=2
  v8i g1;
  g1[0] = (int)(1u << 16);                           // workgroup_mask=0, data_size=1 (2B)
  g1[1] = (int)((dim0 & 0xFFFFu) << 16);             // tensor_dim0[15:0] @ bit48
  g1[2] = (int)((dim0 >> 16) | ((dim1 & 0xFFFFu) << 16));   // dim0[31:16], dim1[15:0]
  g1[3] = (int)((dim1 >> 16) | (tile0 << 16));       // dim1[31:16], tile_dim0
  g1[4] = (int)(tile1 & 0xFFFFu);                    // tile_dim1, tile_dim2=0
  g1[5] = (int)stride0;                              // tensor_dim0_stride[31:0]
  g1[6] = 0;                                         // stride0[47:32]=0, stride1=0
  g1[7] = 0;
  v4i z4 = {};
  v8i z8 = {};
  __builtin_amdgcn_tensor_load_to_lds(g0, g1, z4, z4, z8, 0);
}
#endif

// Manual bf16 tile copy (used for gathered rows; fallback when no TDM builtin).
// dst = [128][BK] row-major; src row-major with leading dim ld; zero-fills OOB.
__device__ __forceinline__ void stage_bf16_tile(
    unsigned short* __restrict__ dst,
    const unsigned short* __restrict__ src, int ld,
    const int* __restrict__ ridx, int row0, int nrows,
    int k0, int K, int tid)
{
  int r  = tid >> 1;
  int cb = (tid & 1) * 32;
  long gr = -1;
  int grow = row0 + r;
  if (grow < nrows) gr = ridx ? (long)ridx[grow] : (long)grow;
  unsigned short* d = dst + r * BK + cb;
  if (gr >= 0) {
    const unsigned short* s = src + gr * (long)ld + k0 + cb;
#pragma unroll
    for (int c = 0; c < 32; c += 8) {
      int gk = k0 + cb + c;
      if (gk + 7 < K) {
        *(v4u*)(d + c) = *(const v4u*)(s + c);
      } else {
#pragma unroll
        for (int t = 0; t < 8; ++t)
          d[c + t] = (gk + t < K) ? s[c + t] : (unsigned short)0;
      }
    }
    if (k0 + BK < K) __builtin_prefetch(s + BK, 0, 0);
  } else {
    v4u z = {};
#pragma unroll
    for (int c = 0; c < 32; c += 8) *(v4u*)(d + c) = z;
  }
}

// ---------------------------------------------------------------------------
// WMMA GEMM:  C[N,M] (ldc) = gather(A_bf16)[N,K] (lda) @ Wt_bf16[M,K] (ldwt)^T
//             (+ bias) (+= C) (ReLU) (bf16 or f32 out)
// Tiles DMA'd into LDS by the Tensor Data Mover when rows are not gathered.
// ---------------------------------------------------------------------------
__global__ __launch_bounds__(256) void gemm_wmma_kernel(
    const unsigned short* __restrict__ A, int lda, const int* __restrict__ arow,
    const unsigned short* __restrict__ Wt, int ldwt, const float* __restrict__ bias,
    void* __restrict__ Cv, int ldc, int N, int K, int M, int flags)
{
  __shared__ unsigned short lds_a[BM * BK];          // [row][k]
  __shared__ unsigned short lds_b[BN * BK];          // [col][k]

  const int tid  = threadIdx.x;
  const int lane = tid & 31;
  const int wave = tid >> 5;
  const int wm   = wave >> 1;
  const int wn   = wave & 1;
  const int row0 = blockIdx.x * BM;
  const int col0 = blockIdx.y * BN;
  const int lh   = lane & 15;
  const int khi  = (lane >> 4) * 8;

  v8f acc[2][4] = {};

  for (int k0 = 0; k0 < K; k0 += BK) {
#ifdef HAVE_TDM
    const bool a_tdm = (arow == nullptr);
    if (wave == 0) {
      asm volatile("s_wait_dscnt 0" ::: "memory");   // LDS reads done before DMA overwrite
      if (a_tdm)
        tdm_load_2d((unsigned)(unsigned long long)&lds_a[0],
                    A + (size_t)row0 * lda + k0,
                    (unsigned)(K - k0), (unsigned)(N - row0), BK, BM, (unsigned)lda);
      tdm_load_2d((unsigned)(unsigned long long)&lds_b[0],
                  Wt + (size_t)col0 * ldwt + k0,
                  (unsigned)(K - k0), (unsigned)(M - col0), BK, BN, (unsigned)ldwt);
    }
    if (!a_tdm)
      stage_bf16_tile(lds_a, A, lda, arow, row0, N, k0, K, tid);
    if (wave == 0)
      __builtin_amdgcn_s_wait_tensorcnt(0);
#else
    stage_bf16_tile(lds_a, A, lda, arow, row0, N, k0, K, tid);
    stage_bf16_tile(lds_b, Wt, ldwt, nullptr, col0, M, k0, K, tid);
#endif
    __syncthreads();

#pragma unroll
    for (int kk = 0; kk < 2; ++kk) {
      const int kb = kk * 32 + khi;   // lanes 0-15: K 0-7/16-23; lanes 16-31: 8-15/24-31
      Frag a0, a1, b[4];
      a0.u[0] = *(const v4u*)&lds_a[(wm * 32 + 0  + lh) * BK + kb];
      a0.u[1] = *(const v4u*)&lds_a[(wm * 32 + 0  + lh) * BK + kb + 16];
      a1.u[0] = *(const v4u*)&lds_a[(wm * 32 + 16 + lh) * BK + kb];
      a1.u[1] = *(const v4u*)&lds_a[(wm * 32 + 16 + lh) * BK + kb + 16];
#pragma unroll
      for (int j = 0; j < 4; ++j) {
        b[j].u[0] = *(const v4u*)&lds_b[(wn * 64 + j * 16 + lh) * BK + kb];
        b[j].u[1] = *(const v4u*)&lds_b[(wn * 64 + j * 16 + lh) * BK + kb + 16];
      }
#pragma unroll
      for (int j = 0; j < 4; ++j) {
        acc[0][j] = __builtin_amdgcn_wmma_f32_16x16x32_bf16(
            false, a0.v, false, b[j].v, (short)0, acc[0][j], false, false);
        acc[1][j] = __builtin_amdgcn_wmma_f32_16x16x32_bf16(
            false, a1.v, false, b[j].v, (short)0, acc[1][j], false, false);
      }
    }
    __syncthreads();
  }

  float* Cf = (float*)Cv;
  unsigned short* Cb = (unsigned short*)Cv;
#pragma unroll
  for (int i = 0; i < 2; ++i) {
#pragma unroll
    for (int j = 0; j < 4; ++j) {
      int c = col0 + wn * 64 + j * 16 + lh;
      if (c >= M) continue;
      float bsum = bias ? bias[c] : 0.f;
#pragma unroll
      for (int v = 0; v < 8; ++v) {
        int r = row0 + wm * 32 + i * 16 + khi + v;
        if (r >= N) continue;
        float val = acc[i][j][v] + bsum;
        if (flags & GF_ACCUM) val += Cf[(long)r * ldc + c];
        if (flags & GF_RELU)  val = fmaxf(val, 0.f);
        if (flags & GF_OUTBF) Cb[(long)r * ldc + c] = f2bf(val);
        else                  Cf[(long)r * ldc + c] = val;
      }
    }
  }
}

// ---------------------------------------------------------------------------
// Conversion kernels (run once per call; all GEMM operands live in bf16)
// ---------------------------------------------------------------------------
__global__ void cvt_bf16_kernel(const float* __restrict__ s, unsigned short* __restrict__ d, long n) {
  long i = (long)blockIdx.x * blockDim.x + threadIdx.x;
  if (i < n) d[i] = f2bf(s[i]);
}

// W[K,M] f32 -> Wt[M,K] bf16 (transposed so GEMM B tiles are row-major copies)
__global__ void wt_convert_kernel(const float* __restrict__ W, unsigned short* __restrict__ Wt,
                                  int K, int M) {
  long i = (long)blockIdx.x * blockDim.x + threadIdx.x;
  if (i < (long)K * M) {
    int k = (int)(i / M), m = (int)(i % M);
    Wt[(long)m * K + k] = f2bf(W[i]);
  }
}

// ---------------------------------------------------------------------------
// GATv2 edge kernels (HEADS=4, CH=128)
// ---------------------------------------------------------------------------
__global__ void init_seg_kernel(float* __restrict__ segmax, float* __restrict__ denom, int n) {
  int i = blockIdx.x * blockDim.x + threadIdx.x;
  if (i < n) { segmax[i] = -__builtin_inff(); denom[i] = 0.f; }
}

__global__ void init_acc_kernel(float* __restrict__ acc, const float* __restrict__ b0,
                                const float* __restrict__ b1, const float* __restrict__ b2,
                                long n) {
  long i = (long)blockIdx.x * blockDim.x + threadIdx.x;
  if (i < n) {
    int c = (int)(i & (HID - 1));
    float v = b0[c];
    if (b1) v += b1[c];
    if (b2) v += b2[c];
    acc[i] = v;
  }
}

__global__ void iota_kernel(int* __restrict__ p, int n) {
  int i = blockIdx.x * blockDim.x + threadIdx.x;
  if (i < n) p[i] = i;
}

__global__ __launch_bounds__(256) void edge_logits_kernel(
    const float* __restrict__ xl, const float* __restrict__ xr,
    const int* __restrict__ src, const int* __restrict__ dst, int E,
    const float* __restrict__ att,
    float* __restrict__ logits, float* __restrict__ segmax)
{
  int item = blockIdx.x * 8 + (threadIdx.x >> 5);
  int lane = threadIdx.x & 31;
  if (item >= E * HEADS) return;
  int e = item >> 2, h = item & 3;
  int si = src[e], di = dst[e];
  float4 a = *(const float4*)(xl + (long)si * HID + h * CH + lane * 4);
  float4 b = *(const float4*)(xr + (long)di * HID + h * CH + lane * 4);
  float4 t;
  t.x = a.x + b.x; t.y = a.y + b.y; t.z = a.z + b.z; t.w = a.w + b.w;
  t.x = t.x > 0.f ? t.x : NEG_SLOPE * t.x;
  t.y = t.y > 0.f ? t.y : NEG_SLOPE * t.y;
  t.z = t.z > 0.f ? t.z : NEG_SLOPE * t.z;
  t.w = t.w > 0.f ? t.w : NEG_SLOPE * t.w;
  float4 w = *(const float4*)(att + h * CH + lane * 4);
  float s = t.x * w.x + t.y * w.y + t.z * w.z + t.w * w.w;
#pragma unroll
  for (int m = 16; m; m >>= 1) s += __shfl_xor(s, m);
  if (lane == 0) {
    logits[item] = s;
    float* mp = &segmax[di * HEADS + h];
    float old = *mp;
    while (s > old) {
      unsigned int assumed = __float_as_uint(old);
      unsigned int got = atomicCAS((unsigned int*)mp, assumed, __float_as_uint(s));
      if (got == assumed) break;
      old = __uint_as_float(got);
    }
  }
}

__global__ void edge_exp_kernel(float* __restrict__ logits, const int* __restrict__ dst, int E,
                                const float* __restrict__ segmax, float* __restrict__ denom)
{
  int item = blockIdx.x * blockDim.x + threadIdx.x;
  if (item >= E * HEADS) return;
  int e = item >> 2, h = item & 3;
  int d = dst[e] * HEADS + h;
  float ex = __expf(logits[item] - segmax[d]);
  logits[item] = ex;
  atomicAdd(&denom[d], ex);
}

__global__ __launch_bounds__(256) void edge_scatter_kernel(
    const float* __restrict__ xl, const int* __restrict__ src, const int* __restrict__ dst,
    int E, const float* __restrict__ logits, const float* __restrict__ denom,
    float* __restrict__ acc)
{
  int item = blockIdx.x * 8 + (threadIdx.x >> 5);
  int lane = threadIdx.x & 31;
  if (item >= E * HEADS) return;
  int e = item >> 2, h = item & 3;
  int si = src[e], di = dst[e];
  float alpha = logits[item] / fmaxf(denom[di * HEADS + h], 1e-16f);
  float4 a = *(const float4*)(xl + (long)si * HID + h * CH + lane * 4);
  float* p = acc + (long)di * HID + h * CH + lane * 4;
  atomicAdd(p + 0, alpha * a.x);
  atomicAdd(p + 1, alpha * a.y);
  atomicAdd(p + 2, alpha * a.z);
  atomicAdd(p + 3, alpha * a.w);
}

// acc f32 -> relu(acc*invR) as bf16 feature buffer for the next GEMMs
__global__ void finalize_kernel(const float* __restrict__ acc, unsigned short* __restrict__ out,
                                long n, float invR) {
  long i = (long)blockIdx.x * blockDim.x + threadIdx.x;
  if (i < n) out[i] = f2bf(fmaxf(acc[i] * invR, 0.f));
}

// ---------------------------------------------------------------------------
// Scoring head kernels (DOT = 256). One wave per row.
// ---------------------------------------------------------------------------
__global__ __launch_bounds__(256) void layernorm_kernel(
    float* __restrict__ X, const float* __restrict__ g, const float* __restrict__ b, int N)
{
  int row = blockIdx.x * 8 + (threadIdx.x >> 5);
  int lane = threadIdx.x & 31;
  if (row >= N) return;
  float* p = X + (long)row * DOT + lane * 8;
  float4 v0 = *(const float4*)p;
  float4 v1 = *(const float4*)(p + 4);
  float s = v0.x + v0.y + v0.z + v0.w + v1.x + v1.y + v1.z + v1.w;
#pragma unroll
  for (int m = 16; m; m >>= 1) s += __shfl_xor(s, m);
  float mu = s * (1.f / DOT);
  float d0 = v0.x - mu, d1 = v0.y - mu, d2 = v0.z - mu, d3 = v0.w - mu;
  float d4 = v1.x - mu, d5 = v1.y - mu, d6 = v1.z - mu, d7 = v1.w - mu;
  float q = d0*d0 + d1*d1 + d2*d2 + d3*d3 + d4*d4 + d5*d5 + d6*d6 + d7*d7;
#pragma unroll
  for (int m = 16; m; m >>= 1) q += __shfl_xor(q, m);
  float rs = rsqrtf(q * (1.f / DOT) + 1e-5f);
  const float* gp = g + lane * 8;
  const float* bp = b + lane * 8;
  float4 g0 = *(const float4*)gp, g1 = *(const float4*)(gp + 4);
  float4 b0 = *(const float4*)bp, b1 = *(const float4*)(bp + 4);
  v0.x = d0 * rs * g0.x + b0.x; v0.y = d1 * rs * g0.y + b0.y;
  v0.z = d2 * rs * g0.z + b0.z; v0.w = d3 * rs * g0.w + b0.w;
  v1.x = d4 * rs * g1.x + b1.x; v1.y = d5 * rs * g1.y + b1.y;
  v1.z = d6 * rs * g1.z + b1.z; v1.w = d7 * rs * g1.w + b1.w;
  *(float4*)p = v0;
  *(float4*)(p + 4) = v1;
}

__global__ __launch_bounds__(256) void dot_kernel(
    const float* __restrict__ Q, const float* __restrict__ R,
    const float* __restrict__ sbias, float* __restrict__ out, int N)
{
  int row = blockIdx.x * 8 + (threadIdx.x >> 5);
  int lane = threadIdx.x & 31;
  if (row >= N) return;
  const float* q = Q + (long)row * DOT + lane * 8;
  const float* r = R + (long)row * DOT + lane * 8;
  float4 q0 = *(const float4*)q, q1 = *(const float4*)(q + 4);
  float4 r0 = *(const float4*)r, r1 = *(const float4*)(r + 4);
  float s = q0.x*r0.x + q0.y*r0.y + q0.z*r0.z + q0.w*r0.w +
            q1.x*r1.x + q1.y*r1.y + q1.z*r1.z + q1.w*r1.w;
#pragma unroll
  for (int m = 16; m; m >>= 1) s += __shfl_xor(s, m);
  if (lane == 0) out[row] = s * (1.f / 16.f) + sbias[0];   // 1/sqrt(256)
}

// ---------------------------------------------------------------------------
// Host-side driver
// ---------------------------------------------------------------------------
static void gemm(hipStream_t s, const unsigned short* A, int lda, const int* arow,
                 const unsigned short* Wt, int ldwt, const float* bias,
                 void* C, int ldc, int N, int K, int M, int flags) {
  dim3 grid((N + BM - 1) / BM, (M + BN - 1) / BN);
  gemm_wmma_kernel<<<grid, dim3(256), 0, s>>>(A, lda, arow, Wt, ldwt, bias, C, ldc, N, K, M, flags);
}

struct GatParams { const float *Wl, *Wr, *att, *bias, *bl, *br; };
struct GatWt { const unsigned short *Wl, *Wr; };

// pytree leaf order (nested dicts alphabetical): af1{W,b} af2{W,b}
// layers[0..1]{comp,gen,leads,revgen,revleads,sim}{Wl,Wr,att,bias,bl,br}
// proj_a proj_q proj_t qh qh_ln{b,g} rh rh_ln{b,g} score_bias
#define PB 11
enum { REL_COMP = 0, REL_GEN, REL_LEADS, REL_REVGEN, REL_REVLEADS, REL_SIM };

static GatParams gat_params(void* const* d_in, int layer, int rel) {
  int base = PB + 4 + layer * 36 + rel * 6;
  GatParams p;
  p.Wl   = (const float*)d_in[base + 0];
  p.Wr   = (const float*)d_in[base + 1];
  p.att  = (const float*)d_in[base + 2];
  p.bias = (const float*)d_in[base + 3];
  p.bl   = (const float*)d_in[base + 4];
  p.br   = (const float*)d_in[base + 5];
  return p;
}

static void run_gat(hipStream_t s, const GatParams& P, const GatWt& WT,
                    const unsigned short* xsrc, int n_src,
                    const unsigned short* xdst, int n_dst,
                    const int* src, const int* dst, int E,
                    const int* loop_idx, int n_loop,
                    float* acc, float* xl, float* xr,
                    float* logits, float* segmax, float* denom)
{
  gemm(s, xsrc, HID, nullptr, WT.Wl, HID, P.bl, xl, HID, n_src, HID, HID, 0);
  gemm(s, xdst, HID, nullptr, WT.Wr, HID, P.br, xr, HID, n_dst, HID, HID, 0);
  int nseg = n_dst * HEADS;
  init_seg_kernel<<<(nseg + 255) / 256, 256, 0, s>>>(segmax, denom, nseg);
  int items = E * HEADS;
  edge_logits_kernel<<<(items + 7) / 8, 256, 0, s>>>(xl, xr, src, dst, E, P.att, logits, segmax);
  if (loop_idx)
    edge_logits_kernel<<<(n_loop * HEADS + 7) / 8, 256, 0, s>>>(
        xl, xr, loop_idx, loop_idx, n_loop, P.att, logits + (long)E * HEADS, segmax);
  edge_exp_kernel<<<(items + 255) / 256, 256, 0, s>>>(logits, dst, E, segmax, denom);
  if (loop_idx)
    edge_exp_kernel<<<(n_loop * HEADS + 255) / 256, 256, 0, s>>>(
        logits + (long)E * HEADS, loop_idx, n_loop, segmax, denom);
  edge_scatter_kernel<<<(items + 7) / 8, 256, 0, s>>>(xl, src, dst, E, logits, denom, acc);
  if (loop_idx)
    edge_scatter_kernel<<<(n_loop * HEADS + 7) / 8, 256, 0, s>>>(
        xl, loop_idx, loop_idx, n_loop, logits + (long)E * HEADS, denom, acc);
}

extern "C" void kernel_launch(void* const* d_in, const int* in_sizes, int n_in,
                              void* d_out, int out_size, void* d_ws, size_t ws_size,
                              hipStream_t stream)
{
  const float* x_query         = (const float*)d_in[0];
  const float* x_think         = (const float*)d_in[1];
  const float* x_answer        = (const float*)d_in[2];
  const float* answer_features = (const float*)d_in[3];
  const int* ei_gen      = (const int*)d_in[4];
  const int* ei_revgen   = (const int*)d_in[5];
  const int* ei_leads    = (const int*)d_in[6];
  const int* ei_revleads = (const int*)d_in[7];
  const int* ei_sim      = (const int*)d_in[8];
  const int* ei_comp     = (const int*)d_in[9];
  const int* ei_score    = (const int*)d_in[10];

  const int NQ = in_sizes[0] / F_IN;
  const int NT = in_sizes[1] / F_IN;
  const int NA = in_sizes[2] / F_IN;
  const int Egen   = in_sizes[4] / 2;
  const int Esim   = in_sizes[8] / 2;
  const int Ecomp  = in_sizes[9] / 2;
  const int Escore = in_sizes[10] / 2;

  const float* af1W  = (const float*)d_in[PB + 0];
  const float* af1b  = (const float*)d_in[PB + 1];
  const float* af2W  = (const float*)d_in[PB + 2];
  const float* af2b  = (const float*)d_in[PB + 3];
  const int PTAIL = PB + 4 + 72;
  const float* projaW = (const float*)d_in[PTAIL + 0];
  const float* projab = (const float*)d_in[PTAIL + 1];
  const float* projqW = (const float*)d_in[PTAIL + 2];
  const float* projqb = (const float*)d_in[PTAIL + 3];
  const float* projtW = (const float*)d_in[PTAIL + 4];
  const float* projtb = (const float*)d_in[PTAIL + 5];
  const float* qhW    = (const float*)d_in[PTAIL + 6];
  const float* qhb    = (const float*)d_in[PTAIL + 7];
  const float* qhln_b = (const float*)d_in[PTAIL + 8];
  const float* qhln_g = (const float*)d_in[PTAIL + 9];
  const float* rhW    = (const float*)d_in[PTAIL + 10];
  const float* rhb    = (const float*)d_in[PTAIL + 11];
  const float* rhln_b = (const float*)d_in[PTAIL + 12];
  const float* rhln_g = (const float*)d_in[PTAIL + 13];
  const float* sbias  = (const float*)d_in[PTAIL + 14];

  // ---- workspace: f32 region then bf16 region ----
  float* Wf = (float*)d_ws;
  size_t fo = 0;
  auto ftake = [&](size_t n) { float* p = Wf + fo; fo += n; return p; };
  float* q_acc  = ftake((size_t)NQ * HID);
  float* t_acc  = ftake((size_t)NT * HID);
  float* a_acc  = ftake((size_t)NA * HID);
  float* xl_buf = ftake((size_t)32768 * HID);
  float* xr_buf = ftake((size_t)32768 * HID);
  float* logits = ftake((size_t)(Ecomp + NQ) * HEADS);
  float* segmax = ftake((size_t)32768 * HEADS);
  float* denom  = ftake((size_t)32768 * HEADS);
  int*   iota   = (int*)ftake((size_t)NQ);

  unsigned short* Bp = (unsigned short*)(Wf + fo);
  size_t bo = 0;
  auto btake = [&](size_t n) { unsigned short* p = Bp + bo; bo += n; return p; };
  unsigned short* fa_q = btake((size_t)NQ * HID);   // feature set A (bf16)
  unsigned short* fa_t = btake((size_t)NT * HID);
  unsigned short* fa_a = btake((size_t)NA * HID);
  unsigned short* fb_q = btake((size_t)NQ * HID);   // feature set B (bf16)
  unsigned short* fb_t = btake((size_t)NT * HID);
  unsigned short* fb_a = btake((size_t)NA * HID);
  unsigned short* xq_in = btake((size_t)NQ * F_IN);
  unsigned short* xt_in = btake((size_t)NT * F_IN);
  unsigned short* xa_in = btake((size_t)NA * F_IN);
  unsigned short* af_in = btake((size_t)Egen * 8);
  unsigned short* hid64b = btake((size_t)Egen * 64);
  unsigned short* afeb   = btake((size_t)Egen * HID);
  (void)ws_size; (void)n_in; (void)out_size;

  auto cvt = [&](const float* s, unsigned short* d, long n) {
    cvt_bf16_kernel<<<(unsigned)((n + 255) / 256), 256, 0, stream>>>(s, d, n);
  };
  auto wt_convert = [&](const float* Wsrc, int K, int M) -> const unsigned short* {
    unsigned short* p = btake((size_t)K * M);
    long n = (long)K * M;
    wt_convert_kernel<<<(unsigned)((n + 255) / 256), 256, 0, stream>>>(Wsrc, p, K, M);
    return p;
  };

  // ---- one-time per-call conversions (weights transposed to [M,K] bf16) ----
  const unsigned short* WtProjQ = wt_convert(projqW, F_IN, HID);
  const unsigned short* WtProjT = wt_convert(projtW, F_IN, HID);
  const unsigned short* WtProjA = wt_convert(projaW, F_IN, HID);
  const unsigned short* WtAf1   = wt_convert(af1W, 6, 64);
  const unsigned short* WtAf2   = wt_convert(af2W, 64, HID);
  const unsigned short* WtQh    = wt_convert(qhW, HID, DOT);
  const unsigned short* WtRh    = wt_convert(rhW, 3 * HID, DOT);
  GatWt gwt[2][6];
  for (int L = 0; L < 2; ++L)
    for (int rel = 0; rel < 6; ++rel) {
      GatParams p = gat_params(d_in, L, rel);
      gwt[L][rel].Wl = wt_convert(p.Wl, HID, HID);
      gwt[L][rel].Wr = wt_convert(p.Wr, HID, HID);
    }
  cvt(x_query,  xq_in, (long)NQ * F_IN);
  cvt(x_think,  xt_in, (long)NT * F_IN);
  cvt(x_answer, xa_in, (long)NA * F_IN);
  cvt(answer_features, af_in, (long)Egen * 6);
  iota_kernel<<<(NQ + 255) / 256, 256, 0, stream>>>(iota, NQ);

  // ---- input projections (bf16 in, bf16 out) ----
  gemm(stream, xq_in, F_IN, nullptr, WtProjQ, F_IN, projqb, fa_q, HID, NQ, F_IN, HID, GF_OUTBF);
  gemm(stream, xt_in, F_IN, nullptr, WtProjT, F_IN, projtb, fa_t, HID, NT, F_IN, HID, GF_OUTBF);
  gemm(stream, xa_in, F_IN, nullptr, WtProjA, F_IN, projab, fa_a, HID, NA, F_IN, HID, GF_OUTBF);

  // ---- 2 hetero GATv2 layers (ping-pong bf16 feature sets, f32 accumulators) ----
  const unsigned short *cq = fa_q, *ct = fa_t, *ca = fa_a;
  unsigned short *oq = fb_q, *ot = fb_t, *oa = fb_a;
  for (int L = 0; L < 2; ++L) {
    GatParams Pcomp     = gat_params(d_in, L, REL_COMP);
    GatParams Pgen      = gat_params(d_in, L, REL_GEN);
    GatParams Pleads    = gat_params(d_in, L, REL_LEADS);
    GatParams Prevgen   = gat_params(d_in, L, REL_REVGEN);
    GatParams Prevleads = gat_params(d_in, L, REL_REVLEADS);
    GatParams Psim      = gat_params(d_in, L, REL_SIM);

    init_acc_kernel<<<(unsigned)(((size_t)NQ * HID + 255) / 256), 256, 0, stream>>>(
        q_acc, Prevgen.bias, Psim.bias, nullptr, (long)NQ * HID);
    init_acc_kernel<<<(unsigned)(((size_t)NT * HID + 255) / 256), 256, 0, stream>>>(
        t_acc, Pgen.bias, Prevleads.bias, Pcomp.bias, (long)NT * HID);
    init_acc_kernel<<<(unsigned)(((size_t)NA * HID + 255) / 256), 256, 0, stream>>>(
        a_acc, Pleads.bias, nullptr, nullptr, (long)NA * HID);

    // q_new = (revgen(t->q) + sim(q->q, +self loops)) / 2
    run_gat(stream, Prevgen, gwt[L][REL_REVGEN], ct, NT, cq, NQ,
            ei_revgen, ei_revgen + Egen, Egen, nullptr, 0,
            q_acc, xl_buf, xr_buf, logits, segmax, denom);
    run_gat(stream, Psim, gwt[L][REL_SIM], cq, NQ, cq, NQ,
            ei_sim, ei_sim + Esim, Esim, iota, NQ,
            q_acc, xl_buf, xr_buf, logits, segmax, denom);
    // t_new = (gen(q->t) + revleads(a->t) + comp(t->t)) / 3
    run_gat(stream, Pgen, gwt[L][REL_GEN], cq, NQ, ct, NT,
            ei_gen, ei_gen + Egen, Egen, nullptr, 0,
            t_acc, xl_buf, xr_buf, logits, segmax, denom);
    run_gat(stream, Prevleads, gwt[L][REL_REVLEADS], ca, NA, ct, NT,
            ei_revleads, ei_revleads + Egen, Egen, nullptr, 0,
            t_acc, xl_buf, xr_buf, logits, segmax, denom);
    run_gat(stream, Pcomp, gwt[L][REL_COMP], ct, NT, ct, NT,
            ei_comp, ei_comp + Ecomp, Ecomp, nullptr, 0,
            t_acc, xl_buf, xr_buf, logits, segmax, denom);
    // a_new = leads(t->a)
    run_gat(stream, Pleads, gwt[L][REL_LEADS], ct, NT, ca, NA,
            ei_leads, ei_leads + Egen, Egen, nullptr, 0,
            a_acc, xl_buf, xr_buf, logits, segmax, denom);

    finalize_kernel<<<(unsigned)(((size_t)NQ * HID + 255) / 256), 256, 0, stream>>>(
        q_acc, oq, (long)NQ * HID, 0.5f);
    finalize_kernel<<<(unsigned)(((size_t)NT * HID + 255) / 256), 256, 0, stream>>>(
        t_acc, ot, (long)NT * HID, 1.f / 3.f);
    finalize_kernel<<<(unsigned)(((size_t)NA * HID + 255) / 256), 256, 0, stream>>>(
        a_acc, oa, (long)NA * HID, 1.f);

    const unsigned short* t0;
    t0 = cq; cq = oq; oq = (unsigned short*)t0;
    t0 = ct; ct = ot; ot = (unsigned short*)t0;
    t0 = ca; ca = oa; oa = (unsigned short*)t0;
  }

  // ---- scoring head (f32 scratch overlays dead accumulators) ----
  const int* score_q = ei_score;
  const int* score_a = ei_score + Escore;
  float* qproj = t_acc;   // 32768*256 fits
  float* rproj = a_acc;

  gemm(stream, af_in, 6, nullptr, WtAf1, 6, af1b, hid64b, 64, Escore, 6, 64,
       GF_RELU | GF_OUTBF);
  gemm(stream, hid64b, 64, nullptr, WtAf2, 64, af2b, afeb, HID, Escore, 64, HID, GF_OUTBF);

  // q = LN(qh(xq[score_q]))
  gemm(stream, cq, HID, score_q, WtQh, HID, qhb, qproj, DOT, Escore, HID, DOT, 0);
  // r = LN(rh([te, ae, afe])) as 3 K-block GEMMs accumulated (Wt is [DOT, 3*HID])
  gemm(stream, ct,   HID, score_a, WtRh,            3 * HID, rhb,     rproj, DOT, Escore, HID, DOT, 0);
  gemm(stream, ca,   HID, score_a, WtRh + HID,      3 * HID, nullptr, rproj, DOT, Escore, HID, DOT, GF_ACCUM);
  gemm(stream, afeb, HID, nullptr, WtRh + 2 * HID,  3 * HID, nullptr, rproj, DOT, Escore, HID, DOT, GF_ACCUM);

  layernorm_kernel<<<(Escore + 7) / 8, 256, 0, stream>>>(qproj, qhln_g, qhln_b, Escore);
  layernorm_kernel<<<(Escore + 7) / 8, 256, 0, stream>>>(rproj, rhln_g, rhln_b, Escore);

  dot_kernel<<<(Escore + 7) / 8, 256, 0, stream>>>(qproj, rproj, sbias, (float*)d_out, Escore);
}
